// ArcticMoE_20203526160653
// MI455X (gfx1250) — compile-verified, matrix operand order checked by reading, not run on hardware
//
#include <hip/hip_runtime.h>

#define T_TOK 4096
#define HDIM  2048
#define IDIM  1408
#define NEXP  16

typedef __attribute__((ext_vector_type(16))) __bf16          v16bf;
typedef __attribute__((ext_vector_type(16))) unsigned short  v16us;
typedef __attribute__((ext_vector_type(8)))  float           v8f;

// ---- float -> bf16 helpers -------------------------------------------------

// pack two fp32 -> one dword of two bf16 (truncation) with a single v_perm_b32:
// result = { hi[31:16], lo[31:16] }
__device__ __forceinline__ unsigned pack2_bf(float lo, float hi) {
    return __builtin_amdgcn_perm(__builtin_bit_cast(unsigned, hi),
                                 __builtin_bit_cast(unsigned, lo),
                                 0x07060302u);
}

// Load 16 consecutive fp32 (16B aligned) -> bf16 WMMA fragment half-row (8 perms)
__device__ __forceinline__ v16bf load16_cvt(const float* __restrict__ p) {
    union { unsigned u[8]; v16bf v; } cvt;
#pragma unroll
    for (int i = 0; i < 4; i++) {
        float4 x = ((const float4*)p)[i];
        cvt.u[2*i+0] = pack2_bf(x.x, x.y);
        cvt.u[2*i+1] = pack2_bf(x.z, x.w);
    }
    return cvt.v;
}

// Load 16 consecutive bf16 (32B) directly -> fragment (no conversion)
__device__ __forceinline__ v16bf load16_raw(const unsigned short* __restrict__ p) {
    v16us raw = *(const v16us*)p;
    return __builtin_bit_cast(v16bf, raw);
}

// ---------------- zero / init ----------------
__global__ void moe_zero_f32(float* __restrict__ p, int n) {
    int i = blockIdx.x * blockDim.x + threadIdx.x;
    if (i < n) p[i] = 0.0f;
}

__global__ void moe_zero_counts(int* __restrict__ counts) {
    if (threadIdx.x < NEXP) counts[threadIdx.x] = 0;
}

// ---------------- fp32 -> bf16 bulk convert (hidden states), 8 elems/thread --------
__global__ __launch_bounds__(1024) void moe_cvt_bf16(const float* __restrict__ src,
                                                     unsigned short* __restrict__ dst) {
    size_t i = ((size_t)blockIdx.x * blockDim.x + threadIdx.x) * 8;
    float4 a = *(const float4*)(src + i);
    float4 b = *(const float4*)(src + i + 4);
    uint4 o;
    o.x = pack2_bf(a.x, a.y);
    o.y = pack2_bf(a.z, a.w);
    o.z = pack2_bf(b.x, b.y);
    o.w = pack2_bf(b.z, b.w);
    *(uint4*)(dst + i) = o;
}

// ---------------- router: logits -> top-2 -> renormalized weights ----------------
__global__ __launch_bounds__(256) void moe_router(
    const float* __restrict__ hs, const float* __restrict__ gw,
    int* __restrict__ counts, int* __restrict__ list, float* __restrict__ wts)
{
    int t = blockIdx.x * blockDim.x + threadIdx.x;   // exact grid: 16*256 = 4096
    const float* __restrict__ x = hs + (size_t)t * HDIM;

    float acc[NEXP];
#pragma unroll
    for (int e = 0; e < NEXP; e++) acc[e] = 0.0f;

    for (int h = 0; h < HDIM; h += 4) {
        float4 xv = *(const float4*)(x + h);
#pragma unroll
        for (int e = 0; e < NEXP; e++) {
            float4 wv = *(const float4*)(gw + (size_t)e * HDIM + h);
            acc[e] += xv.x * wv.x + xv.y * wv.y + xv.z * wv.z + xv.w * wv.w;
        }
    }

    // top-2 logits; renormalized top-2 softmax weight = sigmoid(l1 - l2)
    float m1 = -3.0e38f; int i1 = 0;
#pragma unroll
    for (int e = 0; e < NEXP; e++) if (acc[e] > m1) { m1 = acc[e]; i1 = e; }
    float m2 = -3.0e38f; int i2 = 0;
#pragma unroll
    for (int e = 0; e < NEXP; e++) if (e != i1 && acc[e] > m2) { m2 = acc[e]; i2 = e; }

    float w1 = __builtin_amdgcn_rcpf(1.0f + __expf(m2 - m1));
    float w2 = 1.0f - w1;

    int s1 = atomicAdd(&counts[i1], 1);
    list[i1 * T_TOK + s1] = t;  wts[i1 * T_TOK + s1] = w1;
    int s2 = atomicAdd(&counts[i2], 1);
    list[i2 * T_TOK + s2] = t;  wts[i2 * T_TOK + s2] = w2;
}

// ---------------- prefix sum over expert counts ----------------
__global__ void moe_prefix(const int* __restrict__ counts, int* __restrict__ offsets) {
    if (threadIdx.x == 0) {
        int s = 0;
        for (int e = 0; e < NEXP; e++) { offsets[e] = s; s += counts[e]; }
        offsets[NEXP] = s;
    }
}

// ---------------- gate/up projection + SiLU*mul, weight folded into h (bf16) --------
// grid: (NEXP, IDIM/128), block: 256 (8 waves, each owns a 16-wide N slice)
__global__ __launch_bounds__(256) void moe_gateup(
    const unsigned short* __restrict__ xb,     // hs pre-converted to bf16 [T, H]
    const float* __restrict__ ws,
    const int* __restrict__ counts, const int* __restrict__ offsets,
    const int* __restrict__ list, const float* __restrict__ wts,
    unsigned short* __restrict__ hbuf)
{
    const int e     = blockIdx.x;
    const int nb    = blockIdx.y * 128 + (threadIdx.x >> 5) * 16;  // column base in I
    const int lane  = threadIdx.x & 31;
    const int count = counts[e];
    const int base  = offsets[e];

    const float* __restrict__ we = ws + (size_t)e * (2 * IDIM) * HDIM;
    const int mrow  = lane & 15;          // A-fragment row
    const int khalf = (lane >> 4) * 16;   // K half-select
    const int ncol  = lane & 15;          // B-fragment column

    const float* __restrict__ grow = we + (size_t)(nb + ncol) * HDIM;
    const float* __restrict__ urow = we + (size_t)(IDIM + nb + ncol) * HDIM;

    for (int m0 = 0; m0 < count; m0 += 16) {
        int mr  = m0 + mrow;
        int tok = (mr < count) ? list[e * T_TOK + mr] : 0;
        const unsigned short* __restrict__ xrow = xb + (size_t)tok * HDIM;

        v8f accg = {}; v8f accu = {};
        for (int k0 = 0; k0 < HDIM; k0 += 32) {
            v16bf a  = load16_raw(xrow + k0 + khalf);     // bf16 activations, no cvt
            v16bf bg = load16_cvt(grow + k0 + khalf);     // fp32 weights -> 8 perms
            v16bf bu = load16_cvt(urow + k0 + khalf);
            accg = __builtin_amdgcn_wmma_f32_16x16x32_bf16(false, a, false, bg,
                                                           (short)0, accg, false, false);
            accu = __builtin_amdgcn_wmma_f32_16x16x32_bf16(false, a, false, bu,
                                                           (short)0, accu, false, false);
        }

        // D layout: VGPR r, lane l -> row = r + 8*(l>=16), col = l&15
#pragma unroll
        for (int r = 0; r < 8; r++) {
            int m  = r + (lane >> 4) * 8;
            int gm = m0 + m;
            if (gm < count) {
                float w = wts[e * T_TOK + gm];
                float g = accg[r], u = accu[r];
                float sig = __builtin_amdgcn_rcpf(1.0f + __expf(-g));
                float h = g * sig * u * w;                       // silu(g)*u*route_w
                unsigned hb = __builtin_bit_cast(unsigned, h);
                hbuf[(size_t)(base + gm) * IDIM + nb + ncol] =
                    (unsigned short)(hb >> 16);                  // bf16 (truncate)
            }
        }
    }
}

// ---------------- down projection: out += h @ w2^T (atomic scatter-add) ----------------
// grid: (NEXP, HDIM/128), block: 256
__global__ __launch_bounds__(256) void moe_down(
    const unsigned short* __restrict__ hbuf, const float* __restrict__ w2s,
    const int* __restrict__ counts, const int* __restrict__ offsets,
    const int* __restrict__ list, float* __restrict__ out)
{
    const int e     = blockIdx.x;
    const int nb    = blockIdx.y * 128 + (threadIdx.x >> 5) * 16;  // column base in H
    const int lane  = threadIdx.x & 31;
    const int count = counts[e];
    const int base  = offsets[e];

    const float* __restrict__ w2 = w2s + (size_t)e * HDIM * IDIM;
    const int mrow  = lane & 15;
    const int khalf = (lane >> 4) * 16;
    const int ncol  = lane & 15;

    const float* __restrict__ brow = w2 + (size_t)(nb + ncol) * IDIM;

    for (int m0 = 0; m0 < count; m0 += 16) {
        int mr   = m0 + mrow;
        int hrow = base + ((mr < count) ? mr : 0);
        const unsigned short* __restrict__ arow = hbuf + (size_t)hrow * IDIM;

        v8f acc = {};
        for (int k0 = 0; k0 < IDIM; k0 += 32) {
            v16bf a = load16_raw(arow + k0 + khalf);      // already bf16, no cvt
            v16bf b = load16_cvt(brow + k0 + khalf);      // fp32 weights -> 8 perms
            acc = __builtin_amdgcn_wmma_f32_16x16x32_bf16(false, a, false, b,
                                                          (short)0, acc, false, false);
        }

#pragma unroll
        for (int r = 0; r < 8; r++) {
            int m  = r + (lane >> 4) * 8;
            int gm = m0 + m;
            if (gm < count) {
                int tok = list[e * T_TOK + gm];
                unsafeAtomicAdd(&out[(size_t)tok * HDIM + nb + ncol], acc[r]);
            }
        }
    }
}

extern "C" void kernel_launch(void* const* d_in, const int* in_sizes, int n_in,
                              void* d_out, int out_size, void* d_ws, size_t ws_size,
                              hipStream_t stream) {
    const float* hs  = (const float*)d_in[0];   // [T, H]
    const float* gw  = (const float*)d_in[1];   // [E, H]
    const float* ws  = (const float*)d_in[2];   // [E, 2I, H]
    const float* w2s = (const float*)d_in[3];   // [E, H, I]
    float* out = (float*)d_out;

    // workspace carve-up
    char*           wsb     = (char*)d_ws;
    int*            counts  = (int*)(wsb);                                 // 16 ints
    int*            offsets = (int*)(wsb + 64);                            // 17 ints
    int*            list    = (int*)(wsb + 256);                           // E*T ints
    float*          wtsv    = (float*)(wsb + 256 + (size_t)NEXP*T_TOK*4);  // E*T floats
    unsigned short* hbuf    = (unsigned short*)(wsb + (1u << 20));         // 2T*I bf16 (~22 MB)
    unsigned short* hs_bf   = (unsigned short*)(wsb + (32u << 20));        // T*H bf16 (~17 MB)

    moe_zero_f32<<<(out_size + 1023) / 1024, 1024, 0, stream>>>(out, out_size);
    moe_zero_counts<<<1, 32, 0, stream>>>(counts);
    moe_cvt_bf16<<<(T_TOK * HDIM) / (1024 * 8), 1024, 0, stream>>>(hs, hs_bf);
    moe_router<<<T_TOK / 256, 256, 0, stream>>>(hs, gw, counts, list, wtsv);
    moe_prefix<<<1, 32, 0, stream>>>(counts, offsets);
    moe_gateup<<<dim3(NEXP, IDIM / 128), 256, 0, stream>>>(hs_bf, ws, counts, offsets,
                                                           list, wtsv, hbuf);
    moe_down<<<dim3(NEXP, HDIM / 128), 256, 0, stream>>>(hbuf, w2s, counts, offsets,
                                                         list, out);
}
